// ChemicalReactionModelV2_42606075576806
// MI455X (gfx1250) — matrix operand order repaired and checked by reading, not compiled, not run on hardware
//
#include <hip/hip_runtime.h>
#include <hip/hip_bf16.h>
#include <math.h>

// ---------------------------------------------------------------------------
// Model constants (match reference)
// ---------------------------------------------------------------------------
#define VOCAB 128
#define DM    512
#define NH    8
#define DH    64
#define FF    2048
#define LENC  6
#define LDEC  6
#define RBFK  64
#define BATCH 16
#define NSEQ  256
#define TDEC  128

typedef __attribute__((ext_vector_type(16))) _Float16 v16h;
typedef __attribute__((ext_vector_type(4)))  _Float16 v4h;
typedef __attribute__((ext_vector_type(8)))  float    v8f;

static __device__ inline v8f wmma16(v16h a, v16h b, v8f c) {
  // D = A(16x32,f16) * B(32x16,f16) + C(16x16,f32)
  return __builtin_amdgcn_wmma_f32_16x16x32_f16(false, a, false, b, (short)0, c,
                                                false, false);
}

static __device__ inline float gelu_f(float x) {
  float x3 = x * x * x;
  return 0.5f * x * (1.f + tanhf(0.79788456080286535588f * (x + 0.044715f * x3)));
}

// A-fragment K index for vgpr-slot v, element j, lane-half hh (16x32 f16 layout)
static __device__ inline int ka_idx(int v, int j, int hh) {
  return hh * 16 + 2 * v + j;
}
// B-fragment K index (32x16 f16 layout)
static __device__ inline int kb_idx(int v, int j, int hh) {
  return hh * 16 + 2 * v + j;
}

// ---------------------------------------------------------------------------
// Weight pre-swizzle: W[K x N] f32 -> f16 B-fragments, layout
//   Wp[((kt*ctn + ct)*32 + lane)*16 + e],  k = kt*32 + hh*16 + e,
//   col = ct*16 + (lane&15), hh = lane>>4.  Columns >= Nn are zero-padded.
// ---------------------------------------------------------------------------
__global__ __launch_bounds__(128)
void pack_w_kernel(const float* __restrict__ W, _Float16* __restrict__ Wp,
                   int Nn, int Nnp, int Kd)
{
  const size_t tid = (size_t)blockIdx.x * 128 + threadIdx.x;   // (kt,ct,lane)
  const int lane = (int)(tid & 31);
  const size_t g = tid >> 5;
  const int ctn = Nnp >> 4;
  const int ct = (int)(g % ctn);
  const int kt = (int)(g / ctn);
  if (kt >= (Kd >> 5)) return;
  const int hh  = lane >> 4;
  const int col = ct * 16 + (lane & 15);
  _Float16* o = Wp + tid * 16;
#pragma unroll
  for (int e = 0; e < 16; ++e) {
    const int k = kt * 32 + hh * 16 + e;
    const float val = (col < Nn) ? W[(size_t)k * Nn + col] : 0.f;
    o[e] = (_Float16)val;
  }
}

// ---------------------------------------------------------------------------
// K pack for scores (B = K^T [64 x Tk] per (b,h)):
//   Kp[(((bh*2 + kt)*ctn + ct)*32 + lane)*16 + e], key = ct*16+(lane&15),
//   feature = kt*32 + hh*16 + e  (16 consecutive features -> b128 loads).
// ---------------------------------------------------------------------------
__global__ __launch_bounds__(128)
void pack_k_kernel(const float* __restrict__ Km, _Float16* __restrict__ Kp, int Tk)
{
  const size_t tid = (size_t)blockIdx.x * 128 + threadIdx.x;
  const int lane = (int)(tid & 31);
  const size_t g = tid >> 5;
  const int ctn = Tk >> 4;
  const int ct = (int)(g % ctn);
  const int kt = (int)((g / ctn) & 1);
  const int bh = (int)(g / (2 * ctn));
  if (bh >= BATCH * NH) return;
  const int b = bh / NH, h = bh % NH;
  const int hh  = lane >> 4;
  const int key = ct * 16 + (lane & 15);
  const float* src = &Km[((size_t)b * Tk + key) * DM + h * DH + kt * 32 + hh * 16];
  _Float16* o = Kp + tid * 16;
#pragma unroll
  for (int e = 0; e < 16; ++e) o[e] = (_Float16)src[e];
}

// ---------------------------------------------------------------------------
// V pack for AV (B = V [Tk x 64] per (b,h)):
//   Vp[(((bh*ktn + kt)*4 + ct)*32 + lane)*16 + e], feature = ct*16+(lane&15),
//   key = kt*32 + hh*16 + e  (stride DM, coalesced across lanes).
// ---------------------------------------------------------------------------
__global__ __launch_bounds__(128)
void pack_v_kernel(const float* __restrict__ Vm, _Float16* __restrict__ Vp, int Tk)
{
  const size_t tid = (size_t)blockIdx.x * 128 + threadIdx.x;
  const int lane = (int)(tid & 31);
  const size_t g = tid >> 5;
  const int ktn = Tk >> 5;
  const int ct = (int)(g & 3);
  const int kt = (int)((g >> 2) % ktn);
  const int bh = (int)((g >> 2) / ktn);
  if (bh >= BATCH * NH) return;
  const int b = bh / NH, h = bh % NH;
  const int hh  = lane >> 4;
  const int col = ct * 16 + (lane & 15);
  _Float16* o = Vp + tid * 16;
#pragma unroll
  for (int e = 0; e < 16; ++e) {
    const int key = kt * 32 + hh * 16 + e;
    o[e] = (_Float16)Vm[((size_t)b * Tk + key) * DM + h * DH + col];
  }
}

// ---------------------------------------------------------------------------
// GEMM: C[M,N] = act(A[M,K] @ W[K,N] + bias[N]); A fp32, W pre-swizzled f16.
// Block = 128 threads (4 waves) -> 16 x 256 output strip, 4 accs per wave.
// ---------------------------------------------------------------------------
__global__ __launch_bounds__(128)
void gemm_wmma_kernel(const float* __restrict__ A, const _Float16* __restrict__ Wp,
                      const float* __restrict__ bias, float* __restrict__ C,
                      int M, int Nn, int Nnp, int Kd, int act)
{
  __shared__ __align__(32) _Float16 sAh[32 * 16];  // lane-major A fragments
  const int lane = threadIdx.x & 31;
  const int wave = threadIdx.x >> 5;
  const int hh   = lane >> 4;
  const int arow = lane & 15;
  const int m0   = blockIdx.y * 16;
  const int ctn  = Nnp >> 4;
  const int ct0  = blockIdx.x * 16 + wave * 4;

  // fill mapping: thread (fr, fc) -> fragment (lanef, e0..e0+3)
  const int fr    = threadIdx.x >> 3;         // 0..15
  const int fc    = (threadIdx.x & 7) * 4;    // 0,4,...,28
  const int lanef = fr + 16 * (fc >> 4);
  const int e0    = fc & 15;

  v8f acc[4] = {{}, {}, {}, {}};
  for (int k0 = 0, kt = 0; k0 < Kd; k0 += 32, ++kt) {
    const float4 t4 =
        *reinterpret_cast<const float4*>(&A[(size_t)(m0 + fr) * Kd + k0 + fc]);
    v4h p;
    p[0] = (_Float16)t4.x; p[1] = (_Float16)t4.y;
    p[2] = (_Float16)t4.z; p[3] = (_Float16)t4.w;
    *reinterpret_cast<v4h*>(&sAh[lanef * 16 + e0]) = p;
    if (k0 + 32 < Kd)  // global_prefetch_b8 for next A tile
      __builtin_prefetch(&A[(size_t)(m0 + fr) * Kd + k0 + 32 + fc], 0, 1);
    __syncthreads();

    const v16h a = *reinterpret_cast<const v16h*>(&sAh[lane * 16]);
#pragma unroll
    for (int nb = 0; nb < 4; ++nb) {
      const int ct  = ct0 + nb;
      const int ctc = (ct < ctn) ? ct : (ctn - 1);
      const v16h b = *reinterpret_cast<const v16h*>(
          &Wp[(((size_t)kt * ctn + ctc) * 32 + lane) * 16]);
      acc[nb] = wmma16(a, b, acc[nb]);
    }
    __syncthreads();
  }

#pragma unroll
  for (int nb = 0; nb < 4; ++nb) {
    const int col = (ct0 + nb) * 16 + arow;
    if (col < Nn) {
      float bv = bias ? bias[col] : 0.f;
#pragma unroll
      for (int r = 0; r < 8; ++r) {
        int row = m0 + r + 8 * hh;
        float v = acc[nb][r] + bv;
        if (act == 1) v = gelu_f(v);
        C[(size_t)row * Nn + col] = v;
      }
    }
  }
}

// ---------------------------------------------------------------------------
// Fused multi-head attention: one wave per (b, h, 16-row q-tile).
// K/V consumed as pre-packed f16 B-fragments (single 32B load each).
// S kept in LDS, softmax in-wave, O = P @ V.
// ---------------------------------------------------------------------------
__global__ __launch_bounds__(32)
void attn_wmma_kernel(const float* __restrict__ Q, const _Float16* __restrict__ Kp,
                      const _Float16* __restrict__ Vp, float* __restrict__ O,
                      const float* __restrict__ bias,
                      const unsigned char* __restrict__ keypad,
                      int Tq, int Tk, int causal)
{
  __shared__ float S[16 * 256];           // score tile, max Tk = 256
  __shared__ float red[32];
  const int lane = threadIdx.x & 31;
  const int hh   = lane >> 4;
  const int arow = lane & 15;
  const int q0   = blockIdx.x * 16;
  const int h    = blockIdx.y;
  const int b    = blockIdx.z;
  const int bh   = b * NH + h;
  const int ctn  = Tk >> 4;
  const int ktn  = Tk >> 5;
  const float scale = 0.125f;             // 1/sqrt(64)

  const float* Qb = Q + ((size_t)b * Tq) * DM + h * DH;

  // Q fragments (k-steps 0 and 32 of dh=64), reused for every key tile
  v16h qa[2];
#pragma unroll
  for (int kk = 0; kk < 2; ++kk)
#pragma unroll
    for (int v = 0; v < 8; ++v) {
      qa[kk][2 * v]     = (_Float16)Qb[(size_t)(q0 + arow) * DM + kk * 32 + ka_idx(v, 0, hh)];
      qa[kk][2 * v + 1] = (_Float16)Qb[(size_t)(q0 + arow) * DM + kk * 32 + ka_idx(v, 1, hh)];
    }

  for (int ct = 0; ct < ctn; ++ct) {
    v8f acc = {};
    const int kcol = ct * 16 + arow;
#pragma unroll
    for (int kk = 0; kk < 2; ++kk) {
      const v16h bf = *reinterpret_cast<const v16h*>(
          &Kp[((((size_t)bh * 2 + kk) * ctn + ct) * 32 + lane) * 16]);
      acc = wmma16(qa[kk], bf, acc);
    }
#pragma unroll
    for (int r = 0; r < 8; ++r) {
      int qrow = r + 8 * hh;
      float s = acc[r] * scale;
      if (bias) s += bias[(((size_t)b * NH + h) * NSEQ + (q0 + qrow)) * NSEQ + kcol];
      if (keypad && keypad[b * Tk + kcol]) s = -1e9f;
      if (causal && kcol > q0 + qrow)      s = -1e9f;
      S[qrow * Tk + kcol] = s;
    }
  }
  __syncthreads();

  // softmax: lane handles row `arow`, half-segment `hh` of [0, Tk)
  const int seg0 = hh * (Tk >> 1);
  const int seg1 = seg0 + (Tk >> 1);
  float mx = -1e30f;
  for (int t = seg0; t < seg1; ++t) mx = fmaxf(mx, S[arow * Tk + t]);
  red[lane] = mx; __syncthreads();
  mx = fmaxf(red[arow], red[arow + 16]); __syncthreads();
  float sum = 0.f;
  for (int t = seg0; t < seg1; ++t) {
    float e = __expf(S[arow * Tk + t] - mx);
    S[arow * Tk + t] = e; sum += e;
  }
  red[lane] = sum; __syncthreads();
  const float inv = 1.f / (red[arow] + red[arow + 16]);
  for (int t = seg0; t < seg1; ++t) S[arow * Tk + t] *= inv;
  __syncthreads();

  float* Ob = O + ((size_t)b * Tq) * DM + h * DH;
#pragma unroll
  for (int nb = 0; nb < 4; ++nb) {        // 4 x 16 = dh columns
    v8f acc = {};
    for (int kt = 0; kt < ktn; ++kt) {
      v16h a;
#pragma unroll
      for (int v = 0; v < 8; ++v) {
        a[2 * v]     = (_Float16)S[arow * Tk + kt * 32 + ka_idx(v, 0, hh)];
        a[2 * v + 1] = (_Float16)S[arow * Tk + kt * 32 + ka_idx(v, 1, hh)];
      }
      const v16h bf = *reinterpret_cast<const v16h*>(
          &Vp[((((size_t)bh * ktn + kt) * 4 + nb) * 32 + lane) * 16]);
      acc = wmma16(a, bf, acc);
    }
    const int vcol = nb * 16 + arow;
#pragma unroll
    for (int r = 0; r < 8; ++r)
      Ob[(size_t)(q0 + r + 8 * hh) * DM + vcol] = acc[r];
  }
}

// ---------------------------------------------------------------------------
// Fused RBF pair-bias (two chained WMMA GEMMs, packed weights).
// ---------------------------------------------------------------------------
__global__ __launch_bounds__(32)
void rbf_bias_kernel(const float* __restrict__ coords,
                     const unsigned char* __restrict__ mask,
                     const _Float16* __restrict__ w1pk, const float* __restrict__ b1p,
                     const _Float16* __restrict__ w2pk, const float* __restrict__ b2p,
                     float* __restrict__ biasOut)
{
  __shared__ float Hs[16 * 64];
  const int lane = threadIdx.x & 31;
  const int hh   = lane >> 4;
  const int arow = lane & 15;
  const size_t r0  = (size_t)blockIdx.x * 16;
  const size_t idx = r0 + arow;
  const int b = (int)(idx >> 16);
  const int i = (int)((idx >> 8) & 255);
  const int j = (int)(idx & 255);

  float dx = coords[((size_t)b * NSEQ + i) * 3 + 0] - coords[((size_t)b * NSEQ + j) * 3 + 0];
  float dy = coords[((size_t)b * NSEQ + i) * 3 + 1] - coords[((size_t)b * NSEQ + j) * 3 + 1];
  float dz = coords[((size_t)b * NSEQ + i) * 3 + 2] - coords[((size_t)b * NSEQ + j) * 3 + 2];
  float dist = sqrtf(dx * dx + dy * dy + dz * dz + 1e-12f);

  const float mustep = 20.0f / 63.0f;     // linspace(0,20,64)
  const float invsig = 64.0f / 20.0f;     // 1/sigma

  v16h ea[2];
#pragma unroll
  for (int kk = 0; kk < 2; ++kk)
#pragma unroll
    for (int v = 0; v < 8; ++v)
#pragma unroll
      for (int jj = 0; jj < 2; ++jj) {
        int k = kk * 32 + ka_idx(v, jj, hh);
        float t = (dist - (float)k * mustep) * invsig;
        ea[kk][2 * v + jj] = (_Float16)__expf(-0.5f * t * t);
      }

  // GEMM1: h = gelu(e @ w1 + b1), w1 is [64,64] (ctn = 4)
#pragma unroll
  for (int nb = 0; nb < 4; ++nb) {
    v8f acc = {};
    const int col = nb * 16 + arow;
#pragma unroll
    for (int kk = 0; kk < 2; ++kk) {
      const v16h bf = *reinterpret_cast<const v16h*>(
          &w1pk[(((size_t)kk * 4 + nb) * 32 + lane) * 16]);
      acc = wmma16(ea[kk], bf, acc);
    }
#pragma unroll
    for (int r = 0; r < 8; ++r)
      Hs[(r + 8 * hh) * RBFK + col] = gelu_f(acc[r] + b1p[col]);
  }
  __syncthreads();

  // GEMM2: bias = h @ w2 + b2, w2 is [64,8] padded to 16 cols (ctn = 1)
  const int col = arow;
  v8f acc = {};
#pragma unroll
  for (int kk = 0; kk < 2; ++kk) {
    v16h a;
#pragma unroll
    for (int v = 0; v < 8; ++v) {
      a[2 * v]     = (_Float16)Hs[arow * RBFK + kk * 32 + ka_idx(v, 0, hh)];
      a[2 * v + 1] = (_Float16)Hs[arow * RBFK + kk * 32 + ka_idx(v, 1, hh)];
    }
    const v16h bf = *reinterpret_cast<const v16h*>(
        &w2pk[((size_t)kk * 32 + lane) * 16]);
    acc = wmma16(a, bf, acc);
  }
  if (col < NH) {
#pragma unroll
    for (int r = 0; r < 8; ++r) {
      size_t ridx = r0 + r + 8 * hh;
      int bb = (int)(ridx >> 16), ii = (int)((ridx >> 8) & 255), jj = (int)(ridx & 255);
      float valid = (!mask[bb * NSEQ + ii] && !mask[bb * NSEQ + jj]) ? 1.f : 0.f;
      biasOut[(((size_t)bb * NH + col) * NSEQ + ii) * NSEQ + jj] = (acc[r] + b2p[col]) * valid;
    }
  }
}

// ---------------------------------------------------------------------------
// Residual add + LayerNorm over D=512; one row per 128-thread block.
// ---------------------------------------------------------------------------
__global__ __launch_bounds__(128)
void add_ln_kernel(const float* __restrict__ X, const float* __restrict__ R,
                   const float* __restrict__ g, const float* __restrict__ bt,
                   float* __restrict__ out)
{
  __shared__ float red[128];
  const size_t row = blockIdx.x;
  const int d4 = threadIdx.x * 4;
  const float4 x4 = *reinterpret_cast<const float4*>(&X[row * DM + d4]);
  const float4 r4 = *reinterpret_cast<const float4*>(&R[row * DM + d4]);
  float4 v4 = make_float4(x4.x + r4.x, x4.y + r4.y, x4.z + r4.z, x4.w + r4.w);
  red[threadIdx.x] = v4.x + v4.y + v4.z + v4.w;
  __syncthreads();
  for (int off = 64; off > 0; off >>= 1) {
    if ((int)threadIdx.x < off) red[threadIdx.x] += red[threadIdx.x + off];
    __syncthreads();
  }
  const float mean = red[0] * (1.f / DM);
  __syncthreads();
  float a0 = v4.x - mean, a1 = v4.y - mean, a2 = v4.z - mean, a3 = v4.w - mean;
  red[threadIdx.x] = a0 * a0 + a1 * a1 + a2 * a2 + a3 * a3;
  __syncthreads();
  for (int off = 64; off > 0; off >>= 1) {
    if ((int)threadIdx.x < off) red[threadIdx.x] += red[threadIdx.x + off];
    __syncthreads();
  }
  const float inv = rsqrtf(red[0] * (1.f / DM) + 1e-5f);
  float4 o4;
  o4.x = a0 * inv * g[d4 + 0] + bt[d4 + 0];
  o4.y = a1 * inv * g[d4 + 1] + bt[d4 + 1];
  o4.z = a2 * inv * g[d4 + 2] + bt[d4 + 2];
  o4.w = a3 * inv * g[d4 + 3] + bt[d4 + 3];
  *reinterpret_cast<float4*>(&out[row * DM + d4]) = o4;
}

// masked mean pool over sequence -> [B, D]
__global__ __launch_bounds__(128)
void pool_kernel(const float* __restrict__ X, const unsigned char* __restrict__ mask,
                 float* __restrict__ gf)
{
  const int b = blockIdx.x;
  float cnt = 0.f;
  for (int n = 0; n < NSEQ; ++n) cnt += mask[b * NSEQ + n] ? 0.f : 1.f;
  float inv = 1.f / (cnt + 1e-6f);
  for (int d = threadIdx.x; d < DM; d += 128) {
    float s = 0.f;
    for (int n = 0; n < NSEQ; ++n)
      if (!mask[b * NSEQ + n]) s += X[((size_t)b * NSEQ + n) * DM + d];
    gf[b * DM + d] = s * inv;
  }
}

// decoder init: dec = product_query + global_feat + sinusoidal PE
__global__ __launch_bounds__(128)
void dec_init_kernel(const float* __restrict__ pq, const float* __restrict__ gf,
                     float* __restrict__ dec)
{
  const int t = blockIdx.x & (TDEC - 1);
  const int b = blockIdx.x >> 7;
  for (int d = threadIdx.x; d < DM; d += 128) {
    int i2 = d >> 1;
    float div = __expf(-(float)(2 * i2) * (9.210340371976184f / (float)DM)); // ln(1e4)
    float pe  = (d & 1) ? __cosf((float)t * div) : __sinf((float)t * div);
    dec[((size_t)b * TDEC + t) * DM + d] = pq[t * DM + d] + gf[b * DM + d] + pe;
  }
}

// embedding gather (one float4 per thread)
__global__ __launch_bounds__(128)
void embed_kernel(const int* __restrict__ ids, const float* __restrict__ emb,
                  float* __restrict__ X)
{
  const size_t row = blockIdx.x;
  const int id = ids[row];
  const int d4 = threadIdx.x * 4;
  *reinterpret_cast<float4*>(&X[row * DM + d4]) =
      *reinterpret_cast<const float4*>(&emb[(size_t)id * DM + d4]);
}

// ---------------------------------------------------------------------------
// Host-side orchestration
// ---------------------------------------------------------------------------
static inline void gemm_p(hipStream_t s, const float* A, const float* W,
                          _Float16* Wp, const float* bias, float* C,
                          int M, int Nn, int Kd, int act)
{
  const int Nnp = (Nn + 15) & ~15;
  const int total = (Kd / 32) * (Nnp / 16) * 32;
  hipLaunchKernelGGL(pack_w_kernel, dim3((total + 127) / 128), dim3(128), 0, s,
                     W, Wp, Nn, Nnp, Kd);
  dim3 g((Nnp + 255) / 256, M / 16);
  hipLaunchKernelGGL(gemm_wmma_kernel, g, dim3(128), 0, s,
                     A, Wp, bias, C, M, Nn, Nnp, Kd, act);
}

static inline void attn_p(hipStream_t s, const float* Q, const float* K,
                          const float* V, float* O, _Float16* Kp, _Float16* Vp,
                          const float* bias, const unsigned char* keypad,
                          int Tq, int Tk, int causal)
{
  const int totK = BATCH * NH * 2 * (Tk / 16) * 32;
  const int totV = BATCH * NH * (Tk / 32) * 4 * 32;
  hipLaunchKernelGGL(pack_k_kernel, dim3((totK + 127) / 128), dim3(128), 0, s, K, Kp, Tk);
  hipLaunchKernelGGL(pack_v_kernel, dim3((totV + 127) / 128), dim3(128), 0, s, V, Vp, Tk);
  hipLaunchKernelGGL(attn_wmma_kernel, dim3(Tq / 16, NH, BATCH), dim3(32), 0, s,
                     Q, Kp, Vp, O, bias, keypad, Tq, Tk, causal);
}

extern "C" void kernel_launch(void* const* d_in, const int* in_sizes, int n_in,
                              void* d_out, int out_size, void* d_ws, size_t ws_size,
                              hipStream_t stream)
{
  (void)in_sizes; (void)n_in; (void)out_size; (void)ws_size;
  // ---- inputs, setup_inputs() insertion order, nested dicts recursively ----
  const int*           r_ids    = (const int*)d_in[0];
  const float*         r_coords = (const float*)d_in[1];
  const unsigned char* r_mask   = (const unsigned char*)d_in[2]; // jax bool = 1B
  const float*         atom_emb = (const float*)d_in[3];
  const float *rbf_w1 = (const float*)d_in[4], *rbf_b1 = (const float*)d_in[5];
  const float *rbf_w2 = (const float*)d_in[6], *rbf_b2 = (const float*)d_in[7];
  const float *e_wq = (const float*)d_in[8],  *e_wk = (const float*)d_in[9];
  const float *e_wv = (const float*)d_in[10], *e_wo = (const float*)d_in[11];
  const float *e_bq = (const float*)d_in[12], *e_bk = (const float*)d_in[13];
  const float *e_bv = (const float*)d_in[14], *e_bo = (const float*)d_in[15];
  const float *e_fw1 = (const float*)d_in[16], *e_fb1 = (const float*)d_in[17];
  const float *e_fw2 = (const float*)d_in[18], *e_fb2 = (const float*)d_in[19];
  const float *e_l1g = (const float*)d_in[20], *e_l1b = (const float*)d_in[21];
  const float *e_l2g = (const float*)d_in[22], *e_l2b = (const float*)d_in[23];
  const float *s_wq = (const float*)d_in[24], *s_wk = (const float*)d_in[25];
  const float *s_wv = (const float*)d_in[26], *s_wo = (const float*)d_in[27];
  const float *s_bq = (const float*)d_in[28], *s_bk = (const float*)d_in[29];
  const float *s_bv = (const float*)d_in[30], *s_bo = (const float*)d_in[31];
  const float *c_wq = (const float*)d_in[32], *c_wk = (const float*)d_in[33];
  const float *c_wv = (const float*)d_in[34], *c_wo = (const float*)d_in[35];
  const float *c_bq = (const float*)d_in[36], *c_bk = (const float*)d_in[37];
  const float *c_bv = (const float*)d_in[38], *c_bo = (const float*)d_in[39];
  const float *d_fw1 = (const float*)d_in[40], *d_fb1 = (const float*)d_in[41];
  const float *d_fw2 = (const float*)d_in[42], *d_fb2 = (const float*)d_in[43];
  const float *d_l1g = (const float*)d_in[44], *d_l1b = (const float*)d_in[45];
  const float *d_l2g = (const float*)d_in[46], *d_l2b = (const float*)d_in[47];
  const float *d_l3g = (const float*)d_in[48], *d_l3b = (const float*)d_in[49];
  const float *prod_q = (const float*)d_in[50];
  const float *h_aw = (const float*)d_in[51], *h_ab = (const float*)d_in[52];
  const float *h_cw = (const float*)d_in[53], *h_cb = (const float*)d_in[54];

  // ---- workspace layout (floats); ~126 MB total ----
  float* ws = (float*)d_ws;
  float* buf_bias = ws;                 ws += (size_t)BATCH * NH * NSEQ * NSEQ;   // 32 MB
  float* buf_x    = ws;                 ws += (size_t)BATCH * NSEQ * DM;          //  8 MB
  float* buf_q    = ws;                 ws += (size_t)BATCH * NSEQ * DM;
  float* buf_k    = ws;                 ws += (size_t)BATCH * NSEQ * DM;
  float* buf_v    = ws;                 ws += (size_t)BATCH * NSEQ * DM;
  float* buf_ao   = ws;                 ws += (size_t)BATCH * NSEQ * DM;
  float* buf_t    = ws;                 ws += (size_t)BATCH * NSEQ * DM;
  float* buf_ff   = ws;                 ws += (size_t)BATCH * NSEQ * FF;          // 32 MB
  float* buf_dec  = ws;                 ws += (size_t)BATCH * TDEC * DM;          //  4 MB
  float* buf_pool = ws;                 ws += (size_t)BATCH * DM;
  _Float16* buf_wp  = (_Float16*)ws;    ws += (size_t)DM * FF / 2;                //  2 MB f16
  _Float16* buf_kp  = (_Float16*)ws;    ws += (size_t)BATCH * NSEQ * DM / 2;      //  4 MB f16
  _Float16* buf_vp  = (_Float16*)ws;    ws += (size_t)BATCH * NSEQ * DM / 2;      //  4 MB f16
  _Float16* buf_w1p = (_Float16*)ws;    ws += (RBFK * RBFK) / 2;
  _Float16* buf_w2p = (_Float16*)ws;    ws += (RBFK * 16) / 2;

  const int ME = BATCH * NSEQ;  // 4096 encoder rows
  const int MD = BATCH * TDEC;  // 2048 decoder rows

  // 1) pack RBF weights, fused RBF pair bias, atom embedding gather
  hipLaunchKernelGGL(pack_w_kernel, dim3(((RBFK / 32) * (RBFK / 16) * 32 + 127) / 128),
                     dim3(128), 0, stream, rbf_w1, buf_w1p, RBFK, RBFK, RBFK);
  hipLaunchKernelGGL(pack_w_kernel, dim3(((RBFK / 32) * 1 * 32 + 127) / 128),
                     dim3(128), 0, stream, rbf_w2, buf_w2p, NH, 16, RBFK);
  hipLaunchKernelGGL(rbf_bias_kernel, dim3((BATCH * NSEQ * NSEQ) / 16), dim3(32),
                     0, stream, r_coords, r_mask, buf_w1p, rbf_b1, buf_w2p, rbf_b2,
                     buf_bias);
  hipLaunchKernelGGL(embed_kernel, dim3(ME), dim3(128), 0, stream,
                     r_ids, atom_emb, buf_x);

  // 2) Encoder stack
  for (int i = 0; i < LENC; ++i) {
    const size_t wdd = (size_t)i * DM * DM, bd = (size_t)i * DM;
    const size_t wdf = (size_t)i * DM * FF, bf = (size_t)i * FF;
    gemm_p(stream, buf_x, e_wq + wdd, buf_wp, e_bq + bd, buf_q, ME, DM, DM, 0);
    gemm_p(stream, buf_x, e_wk + wdd, buf_wp, e_bk + bd, buf_k, ME, DM, DM, 0);
    gemm_p(stream, buf_x, e_wv + wdd, buf_wp, e_bv + bd, buf_v, ME, DM, DM, 0);
    attn_p(stream, buf_q, buf_k, buf_v, buf_ao, buf_kp, buf_vp,
           buf_bias, r_mask, NSEQ, NSEQ, 0);
    gemm_p(stream, buf_ao, e_wo + wdd, buf_wp, e_bo + bd, buf_t, ME, DM, DM, 0);
    hipLaunchKernelGGL(add_ln_kernel, dim3(ME), dim3(128), 0, stream,
                       buf_x, buf_t, e_l1g + bd, e_l1b + bd, buf_x);
    gemm_p(stream, buf_x, e_fw1 + wdf, buf_wp, e_fb1 + bf, buf_ff, ME, FF, DM, 1);
    gemm_p(stream, buf_ff, e_fw2 + wdf, buf_wp, e_fb2 + bd, buf_t, ME, DM, FF, 0);
    hipLaunchKernelGGL(add_ln_kernel, dim3(ME), dim3(128), 0, stream,
                       buf_x, buf_t, e_l2g + bd, e_l2b + bd, buf_x);
  }

  // 3) Pool + decoder init
  hipLaunchKernelGGL(pool_kernel, dim3(BATCH), dim3(128), 0, stream,
                     buf_x, r_mask, buf_pool);
  hipLaunchKernelGGL(dec_init_kernel, dim3(MD), dim3(128), 0, stream,
                     prod_q, buf_pool, buf_dec);

  // 4) Decoder stack
  for (int i = 0; i < LDEC; ++i) {
    const size_t wdd = (size_t)i * DM * DM, bd = (size_t)i * DM;
    const size_t wdf = (size_t)i * DM * FF, bf = (size_t)i * FF;
    // causal self-attention
    gemm_p(stream, buf_dec, s_wq + wdd, buf_wp, s_bq + bd, buf_q, MD, DM, DM, 0);
    gemm_p(stream, buf_dec, s_wk + wdd, buf_wp, s_bk + bd, buf_k, MD, DM, DM, 0);
    gemm_p(stream, buf_dec, s_wv + wdd, buf_wp, s_bv + bd, buf_v, MD, DM, DM, 0);
    attn_p(stream, buf_q, buf_k, buf_v, buf_ao, buf_kp, buf_vp,
           (const float*)nullptr, (const unsigned char*)nullptr, TDEC, TDEC, 1);
    gemm_p(stream, buf_ao, s_wo + wdd, buf_wp, s_bo + bd, buf_t, MD, DM, DM, 0);
    hipLaunchKernelGGL(add_ln_kernel, dim3(MD), dim3(128), 0, stream,
                       buf_dec, buf_t, d_l1g + bd, d_l1b + bd, buf_dec);
    // cross-attention (keys/values from encoder memory buf_x)
    gemm_p(stream, buf_dec, c_wq + wdd, buf_wp, c_bq + bd, buf_q, MD, DM, DM, 0);
    gemm_p(stream, buf_x,  c_wk + wdd, buf_wp, c_bk + bd, buf_k, ME, DM, DM, 0);
    gemm_p(stream, buf_x,  c_wv + wdd, buf_wp, c_bv + bd, buf_v, ME, DM, DM, 0);
    attn_p(stream, buf_q, buf_k, buf_v, buf_ao, buf_kp, buf_vp,
           (const float*)nullptr, r_mask, TDEC, NSEQ, 0);
    gemm_p(stream, buf_ao, c_wo + wdd, buf_wp, c_bo + bd, buf_t, MD, DM, DM, 0);
    hipLaunchKernelGGL(add_ln_kernel, dim3(MD), dim3(128), 0, stream,
                       buf_dec, buf_t, d_l2g + bd, d_l2b + bd, buf_dec);
    // FFN
    gemm_p(stream, buf_dec, d_fw1 + wdf, buf_wp, d_fb1 + bf, buf_ff, MD, FF, DM, 1);
    gemm_p(stream, buf_ff,  d_fw2 + wdf, buf_wp, d_fb2 + bd, buf_t, MD, DM, FF, 0);
    hipLaunchKernelGGL(add_ln_kernel, dim3(MD), dim3(128), 0, stream,
                       buf_dec, buf_t, d_l3g + bd, d_l3b + bd, buf_dec);
  }

  // 5) Output heads -> d_out = [logits(B*128*128) | coords(B*128*3)]
  float* out_logits = (float*)d_out;
  float* out_coords = out_logits + (size_t)MD * VOCAB;
  gemm_p(stream, buf_dec, h_aw, buf_wp, h_ab, out_logits, MD, VOCAB, DM, 0);
  gemm_p(stream, buf_dec, h_cw, buf_wp, h_cb, out_coords, MD, 3, DM, 0);
}
